// AttentionLayer_39840116638061
// MI455X (gfx1250) — matrix operand order repaired and checked by reading, not compiled, not run on hardware
//
#include <hip/hip_runtime.h>
#include <hip/hip_bf16.h>

typedef __attribute__((ext_vector_type(16))) __bf16 v16bf;
typedef __attribute__((ext_vector_type(8)))  __bf16 v8bf;
typedef __attribute__((ext_vector_type(8)))  float  v8f;

#define WMMA_BF16(a, b, c) \
  __builtin_amdgcn_wmma_f32_16x16x32_bf16(false, (a), false, (b), (short)0, (c), false, false)

// ---------------------------------------------------------------------------
// fp32 -> bf16 elementwise conversion (x4); n is a multiple of 1024
// ---------------------------------------------------------------------------
__global__ __launch_bounds__(256) void cvt_f32_bf16(const float* __restrict__ in,
                                                    __bf16* __restrict__ out, int n) {
  int i = (blockIdx.x * 256 + threadIdx.x) * 4;
  if (i + 3 < n) {
    float4 f = *(const float4*)(in + i);
    out[i + 0] = (__bf16)f.x;
    out[i + 1] = (__bf16)f.y;
    out[i + 2] = (__bf16)f.z;
    out[i + 3] = (__bf16)f.w;
  }
}

// ---------------------------------------------------------------------------
// fp32 [K][N] -> bf16 transposed [N][K], 32x32 LDS tiles (one-time weight prep)
// ---------------------------------------------------------------------------
__global__ __launch_bounds__(256) void transpose_cvt(const float* __restrict__ in,
                                                     __bf16* __restrict__ out,
                                                     int K, int N) {
  __shared__ float tile[32][33];
  int tx = threadIdx.x & 31, ty = threadIdx.x >> 5;  // ty: 0..7
  int k0 = blockIdx.y * 32, n0 = blockIdx.x * 32;
#pragma unroll
  for (int i = 0; i < 32; i += 8)
    tile[ty + i][tx] = in[(size_t)(k0 + ty + i) * N + n0 + tx];
  __syncthreads();
#pragma unroll
  for (int i = 0; i < 32; i += 8)
    out[(size_t)(n0 + ty + i) * K + k0 + tx] = (__bf16)tile[tx][ty + i];
}

// ---------------------------------------------------------------------------
// Tiled WMMA GEMM: C[M,N] = A[M,K](bf16 row-major) * BT[N,K](bf16, B transposed)
//                           + bias[N](f32)
// Block tile 128x128, 8 waves (2x4), each wave 64x32 (4x2 WMMA tiles).
// Double-buffered LDS tiles [dim][K=32], stride 40; one barrier / K-step;
// next tile staged in registers while WMMAs consume the current buffer;
// K+2 tile pulled ahead with global_prefetch. All fragments hoisted ahead of
// the 8-WMMA clause so the scheduler can overlap LDS loads with matrix ops.
// MODE 0: f32 out[M][N].
// MODE 2: QKV split: cols <2048 -> qk[row][2048] bf16 ;
//         cols >=2048 -> vT[((b*16+h)*64+hd)][2048] bf16 (V transposed).
// ---------------------------------------------------------------------------
template <int MODE>
__global__ __launch_bounds__(256) void gemm_bias(const __bf16* __restrict__ A,
                                                 const __bf16* __restrict__ BT,
                                                 const float* __restrict__ bias,
                                                 float* __restrict__ outF,
                                                 __bf16* __restrict__ qk,
                                                 __bf16* __restrict__ vT,
                                                 int M, int N, int K) {
  __shared__ __bf16 As[2][128 * 40];
  __shared__ __bf16 Bs[2][128 * 40];

  const int tid  = threadIdx.x;
  const int lane = tid & 31;
  const int wave = tid >> 5;
  const int lan  = lane & 15;
  const int hi   = lane >> 4;
  const int wm   = wave >> 2;   // 0..1
  const int wn   = wave & 3;    // 0..3
  const int m0   = blockIdx.y * 128;
  const int n0   = blockIdx.x * 128;

  v8f zero = {};
  v8f acc[4][2];
#pragma unroll
  for (int i = 0; i < 4; i++)
#pragma unroll
    for (int j = 0; j < 2; j++) acc[i][j] = zero;

  const int r    = tid >> 1;            // 0..127
  const int half = (tid & 1) * 16;      // 0 or 16
  const __bf16* ga = A + (size_t)(m0 + r) * K + half;   // + k0
  const __bf16* gb = BT + (size_t)(n0 + r) * K + half;  // + k0

  // Preload tile 0
  {
    v8bf a0 = *(const v8bf*)ga;
    v8bf a1 = *(const v8bf*)(ga + 8);
    v8bf b0 = *(const v8bf*)gb;
    v8bf b1 = *(const v8bf*)(gb + 8);
    *(v8bf*)&As[0][r * 40 + half]     = a0;
    *(v8bf*)&As[0][r * 40 + half + 8] = a1;
    *(v8bf*)&Bs[0][r * 40 + half]     = b0;
    *(v8bf*)&Bs[0][r * 40 + half + 8] = b1;
  }
  __syncthreads();

  int cur = 0;
  for (int k0 = 0; k0 < K; k0 += 32) {
    const bool have_next = (k0 + 32) < K;
    v8bf na0, na1, nb0, nb1;
    if (have_next) {
      // Stage K+1 tile into registers; pull K+2 ahead into cache
      const __bf16* gan = ga + k0 + 32;
      const __bf16* gbn = gb + k0 + 32;
      na0 = *(const v8bf*)gan;
      na1 = *(const v8bf*)(gan + 8);
      nb0 = *(const v8bf*)gbn;
      nb1 = *(const v8bf*)(gbn + 8);
      if (k0 + 64 < K) {
        __builtin_prefetch(ga + k0 + 64, 0, 1);
        __builtin_prefetch(gb + k0 + 64, 0, 1);
      }
    }

    // Hoist all fragments (6 x 2 b128 LDS loads), then the 8-WMMA clause.
    // B fragments: elem i<8 -> K = i + hi*8 ; i>=8 -> K = 16 + (i-8) + hi*8
    v16bf bf[2];
#pragma unroll
    for (int nt = 0; nt < 2; nt++) {
      int nr = wn * 32 + nt * 16 + lan;
      union { v16bf v; v8bf h2[2]; } u;
      u.h2[0] = *(const v8bf*)&Bs[cur][nr * 40 + hi * 8];
      u.h2[1] = *(const v8bf*)&Bs[cur][nr * 40 + 16 + hi * 8];
      bf[nt] = u.v;
    }
    // A fragments: elem i<8 -> K = i + hi*8 ; i>=8 -> K = 16 + (i-8) + hi*8
    v16bf af[4];
#pragma unroll
    for (int mt = 0; mt < 4; mt++) {
      int mr = wm * 64 + mt * 16 + lan;
      union { v16bf v; v8bf h2[2]; } u;
      u.h2[0] = *(const v8bf*)&As[cur][mr * 40 + hi * 8];
      u.h2[1] = *(const v8bf*)&As[cur][mr * 40 + 16 + hi * 8];
      af[mt] = u.v;
    }
#pragma unroll
    for (int mt = 0; mt < 4; mt++)
#pragma unroll
      for (int nt = 0; nt < 2; nt++)
        acc[mt][nt] = WMMA_BF16(af[mt], bf[nt], acc[mt][nt]);

    if (have_next) {
      // Fill the alternate buffer (other waves may still be reading buf[cur])
      *(v8bf*)&As[cur ^ 1][r * 40 + half]     = na0;
      *(v8bf*)&As[cur ^ 1][r * 40 + half + 8] = na1;
      *(v8bf*)&Bs[cur ^ 1][r * 40 + half]     = nb0;
      *(v8bf*)&Bs[cur ^ 1][r * 40 + half + 8] = nb1;
    }
    __syncthreads();
    cur ^= 1;
  }

  // Epilogue: C layout row = rr + hi*8, col = lan (per 16x16 tile)
#pragma unroll
  for (int mt = 0; mt < 4; mt++) {
#pragma unroll
    for (int nt = 0; nt < 2; nt++) {
      int col = n0 + wn * 32 + nt * 16 + lan;
      float bv = bias[col];
#pragma unroll
      for (int rr = 0; rr < 8; rr++) {
        int row = m0 + wm * 64 + mt * 16 + rr + hi * 8;
        float v = acc[mt][nt][rr] + bv;
        if (MODE == 0) {
          outF[(size_t)row * N + col] = v;
        } else {
          if (col < 2048) {
            qk[(size_t)row * 2048 + col] = (__bf16)v;
          } else {
            int b = row >> 11, s = row & 2047;
            int hh = (col - 2048) >> 6, hd = (col - 2048) & 63;
            vT[(((size_t)b * 16 + hh) * 64 + hd) * 2048 + s] = (__bf16)v;
          }
        }
      }
    }
  }
}

// ---------------------------------------------------------------------------
// Flash attention.
//   qk: [B*S][2048] bf16  (Q at col h*64, K at col 1024+h*64)
//   vT: [(b*16+h)*64+hd][2048] bf16 (V transposed: contiguous over keys)
// One wave per 16-row q-tile, 4 waves/block. Keys in chunks of 32.
// Scores: S(16x32) = Q(16x64) @ K^T(64x32)  -> 4 WMMAs (K frags from global)
// Update: O(16x64) += P(16x32) @ V(32x64)   -> 4 WMMAs (V frags from vT global)
// V fragments are loaded right after the score WMMAs so their latency hides
// under the softmax VALU block and the P LDS round-trip.
// ---------------------------------------------------------------------------
__global__ __launch_bounds__(128) void attn_fa(const __bf16* __restrict__ qk,
                                               const __bf16* __restrict__ vT,
                                               __bf16* __restrict__ ctx) {
  __shared__ __bf16 pbuf[4][16 * 40];  // per-wave P tile [row][key]

  const int tid  = threadIdx.x;
  const int lane = tid & 31;
  const int wave = tid >> 5;
  const int lan  = lane & 15;
  const int hi   = lane >> 4;

  const int bx    = blockIdx.x;        // 2048 blocks = B*H*32 groups
  const int g     = bx & 31;
  const int h     = (bx >> 5) & 15;
  const int b     = bx >> 9;
  const int qtile = g * 4 + wave;      // 0..127
  const int qbase = qtile * 16;

  __bf16* pb = pbuf[wave];

  // Q fragments (A-matrix, 16x64 -> 2 chunks of K=32), reused across all keys
  const __bf16* Qbase = qk + ((size_t)b * 2048 + qbase) * 2048 + h * 64;
  v16bf aq[2];
#pragma unroll
  for (int c = 0; c < 2; c++) {
    const __bf16* qr = Qbase + (size_t)lan * 2048 + c * 32;
    union { v16bf v; v8bf h2[2]; } u;
    u.h2[0] = *(const v8bf*)(qr + hi * 8);
    u.h2[1] = *(const v8bf*)(qr + 16 + hi * 8);
    aq[c] = u.v;
  }

  v8f zero = {};
  v8f o[4];
#pragma unroll
  for (int t = 0; t < 4; t++) o[t] = zero;
  float mrow[8], lrow[8];
#pragma unroll
  for (int r = 0; r < 8; r++) { mrow[r] = -1e30f; lrow[r] = 0.0f; }

  // V^T row base for this (b,h): rows are hd 0..63, contiguous over s
  const __bf16* vbase = vT + (((size_t)b * 16 + h) * 64) * 2048;
  const __bf16* kbase = qk + ((size_t)b * 2048) * 2048 + 1024 + h * 64;

  const int nch = qtile / 2 + 1;  // causal: keys 0 .. qbase+15, 32 at a time
  for (int chn = 0; chn < nch; chn++) {
    const int kcb = chn * 32;

    if (chn + 1 < nch) {
      // Pull next chunk's K rows and V segments toward the WGP
      __builtin_prefetch(kbase + (size_t)(kcb + 32 + lane) * 2048, 0, 1);
      __builtin_prefetch(vbase + (size_t)(lane * 2) * 2048 + kcb + 32, 0, 1);
    }

    // K fragments for both score tiles (8 b128 loads, one clause)
    v16bf kf[2][2];
#pragma unroll
    for (int nt = 0; nt < 2; nt++) {
      const __bf16* kr = kbase + (size_t)(kcb + nt * 16 + lan) * 2048;
#pragma unroll
      for (int c = 0; c < 2; c++) {
        union { v16bf v; v8bf h2[2]; } u;
        u.h2[0] = *(const v8bf*)(kr + c * 32 + hi * 8);
        u.h2[1] = *(const v8bf*)(kr + c * 32 + 16 + hi * 8);
        kf[nt][c] = u.v;
      }
    }
    // Scores: two 16x16 tiles
    v8f s[2];
    s[0] = zero; s[1] = zero;
#pragma unroll
    for (int nt = 0; nt < 2; nt++)
#pragma unroll
      for (int c = 0; c < 2; c++) s[nt] = WMMA_BF16(aq[c], kf[nt][c], s[nt]);

    // V fragments for the P@V step: independent of scores, issue now so the
    // loads complete under the softmax VALU work below.
    v16bf vf[4];
#pragma unroll
    for (int t = 0; t < 4; t++) {
      const __bf16* vr = vbase + (size_t)(t * 16 + lan) * 2048 + kcb;
      union { v16bf v; v8bf h2[2]; } u;
      u.h2[0] = *(const v8bf*)(vr + hi * 8);
      u.h2[1] = *(const v8bf*)(vr + 16 + hi * 8);
      vf[t] = u.v;
    }

    // Causal mask + scale (ref: mask with -1e5 then /8 -> -12500), online softmax
    float pv[2][8];
#pragma unroll
    for (int nt = 0; nt < 2; nt++) {
      int key = kcb + nt * 16 + lan;
#pragma unroll
      for (int r = 0; r < 8; r++) {
        int q = qbase + r + hi * 8;
        pv[nt][r] = (key <= q) ? s[nt][r] * 0.125f : -12500.0f;
      }
    }
#pragma unroll
    for (int r = 0; r < 8; r++) {
      float mx = fmaxf(pv[0][r], pv[1][r]);
#pragma unroll
      for (int msk = 1; msk < 16; msk <<= 1) mx = fmaxf(mx, __shfl_xor(mx, msk, 32));
      float mnew  = fmaxf(mrow[r], mx);
      float scale = __expf(mrow[r] - mnew);
      float p0 = __expf(pv[0][r] - mnew);
      float p1 = __expf(pv[1][r] - mnew);
      pv[0][r] = p0; pv[1][r] = p1;
      float sum = p0 + p1;
#pragma unroll
      for (int msk = 1; msk < 16; msk <<= 1) sum += __shfl_xor(sum, msk, 32);
      lrow[r] = lrow[r] * scale + sum;
      mrow[r] = mnew;
#pragma unroll
      for (int t = 0; t < 4; t++) o[t][r] *= scale;
    }

    // P (C layout) -> LDS -> A-fragment layout
#pragma unroll
    for (int nt = 0; nt < 2; nt++)
#pragma unroll
      for (int r = 0; r < 8; r++)
        pb[(r + hi * 8) * 40 + nt * 16 + lan] = (__bf16)pv[nt][r];

    asm volatile("s_wait_dscnt 0" ::: "memory");  // intra-wave LDS exchange

    v16bf ap;
    {
      union { v16bf v; v8bf h2[2]; } u;
      u.h2[0] = *(const v8bf*)&pb[lan * 40 + hi * 8];
      u.h2[1] = *(const v8bf*)&pb[lan * 40 + 16 + hi * 8];
      ap = u.v;
    }
    // O += P @ V
#pragma unroll
    for (int t = 0; t < 4; t++) o[t] = WMMA_BF16(ap, vf[t], o[t]);
  }

  // Normalize and write ctx[b,s,h*64+hd] (merged heads, bf16)
#pragma unroll
  for (int t = 0; t < 4; t++) {
#pragma unroll
    for (int r = 0; r < 8; r++) {
      int q    = qbase + r + hi * 8;
      int dcol = h * 64 + t * 16 + lan;
      ctx[((size_t)b * 2048 + q) * 1024 + dcol] = (__bf16)(o[t][r] / lrow[r]);
    }
  }
}

// ---------------------------------------------------------------------------
// Launch: cvt X, transpose-cvt weights -> QKV GEMM (split epilogue) ->
//         flash attention -> output projection GEMM
// ---------------------------------------------------------------------------
extern "C" void kernel_launch(void* const* d_in, const int* in_sizes, int n_in,
                              void* d_out, int out_size, void* d_ws, size_t ws_size,
                              hipStream_t stream) {
  const float* hs     = (const float*)d_in[0];
  const float* w_attn = (const float*)d_in[1];
  const float* b_attn = (const float*)d_in[2];
  const float* w_proj = (const float*)d_in[3];
  const float* b_proj = (const float*)d_in[4];

  char* ws = (char*)d_ws;
  __bf16* xbf  = (__bf16*)(ws);                 // [8192][1024]        16 MB
  __bf16* waT  = (__bf16*)(ws + 16777216);      // [3072][1024] (W^T)   6 MB
  __bf16* wpT  = (__bf16*)(ws + 23068672);      // [1024][1024] (W^T)   2 MB
  __bf16* qkb  = (__bf16*)(ws + 25165824);      // [8192][2048]        32 MB
  __bf16* vTb  = (__bf16*)(ws + 58720256);      // [4096][2048] (V^T)  16 MB
  __bf16* ctxb = (__bf16*)(ws + 75497472);      // [8192][1024]        16 MB

  cvt_f32_bf16<<<8192, 256, 0, stream>>>(hs, xbf, 8388608);
  transpose_cvt<<<dim3(96, 32), 256, 0, stream>>>(w_attn, waT, 1024, 3072);
  transpose_cvt<<<dim3(32, 32), 256, 0, stream>>>(w_proj, wpT, 1024, 1024);

  // QKV: [8192,1024] @ [1024,3072] + b_attn -> qk bf16 + V^T bf16
  gemm_bias<2><<<dim3(24, 64), 256, 0, stream>>>(xbf, waT, b_attn, nullptr,
                                                 qkb, vTb, 8192, 3072, 1024);
  // Flash attention: B*H*32 groups x 4 waves
  attn_fa<<<2048, 128, 0, stream>>>(qkb, vTb, ctxb);

  // Out: [8192,1024] @ [1024,1024] + b_proj -> f32 d_out
  gemm_bias<0><<<dim3(8, 64), 256, 0, stream>>>(ctxb, wpT, b_proj, (float*)d_out,
                                                nullptr, nullptr, 8192, 1024, 1024);
}
